// GroupedQueryAttention_90563680403662
// MI455X (gfx1250) — matrix operand order repaired and checked by reading, not compiled, not run on hardware
//
#include <hip/hip_runtime.h>
#include <hip/hip_bf16.h>

// ---------------------------------------------------------------------------
// GQA (32 heads, 8 KV heads, head_dim 64, window 512) on MI455X / gfx1250.
//
// Precision: bf16 operands + fp32 WMMA accumulation (V_WMMA_F32_16X16X32_BF16)
// for all matmuls. ~60 GFLOP over ~100MB unique traffic -> memory floor ~4us
// at 23.3 TB/s; bf16 halves operand bytes and K=32/instr is the deep-K WMMA
// path (fp32 WMMA is K=4 only).
//
// Data movement: projection/O GEMMs are LDS-staged and double-buffered with
// CDNA5 async global->LDS copies (GLOBAL_LOAD_ASYNC_TO_LDS_B128 / ASYNCcnt,
// guarded by __has_builtin with a synchronous LDS fallback). Attention is
// flash-style, one wave per (16-query tile, head), V pre-transposed so PV
// B-fragments are contiguous b128 loads; P goes through LDS to reach the
// A-fragment layout.
// ---------------------------------------------------------------------------

typedef unsigned short u16;
typedef __bf16 v16bf __attribute__((ext_vector_type(16)));
typedef float  v8f   __attribute__((ext_vector_type(8)));

#define T_SEQ    2048
#define DM       2048
#define NH       32
#define NKV      8
#define HD       64
#define KVD      512
#define HALF_WIN 256

// fp32 -> bf16, round-to-nearest-even, pure integer ops
__device__ __forceinline__ u16 f2bf(float x) {
  unsigned u = __float_as_uint(x);
  u += 0x7FFFu + ((u >> 16) & 1u);
  return (u16)(u >> 16);
}

// A-matrix 16x32 bf16 fragment (ISA 7.12.2): lane l holds row m=l&15,
// K = {ka..ka+7, ka+16..ka+23}, ka = (l>>4)*8. Two b128 loads.
__device__ __forceinline__ v16bf load_frag_a(const u16* tile, int ld, int lane) {
  const u16* p = tile + (lane & 15) * ld + ((lane >> 4) << 3);
  union { int4 q[2]; v16bf v; } u;
  u.q[0] = *reinterpret_cast<const int4*>(p);
  u.q[1] = *reinterpret_cast<const int4*>(p + 16);
  return u.v;
}

// B-matrix 32x16 bf16 fragment: lane l holds column n=l&15,
// K = {kb..kb+7, kb+16..kb+23}, kb = (l>>4)*8 (column n of B = row n of W for y=x@W^T).
__device__ __forceinline__ v16bf load_frag_b(const u16* tile, int ld, int lane) {
  const u16* p = tile + (lane & 15) * ld + ((lane >> 4) << 3);
  union { int4 q[2]; v16bf v; } u;
  u.q[0] = *reinterpret_cast<const int4*>(p);
  u.q[1] = *reinterpret_cast<const int4*>(p + 16);
  return u.v;
}

__device__ __forceinline__ v8f wmma_bf16(v16bf a, v16bf b, v8f c) {
  // (neg_a, A, neg_b, B, c_mod, C, reuse_a, reuse_b)
  return __builtin_amdgcn_wmma_f32_16x16x32_bf16(false, a, false, b, (short)0, c,
                                                 false, false);
}

// ---- CDNA5 async global->LDS copy (ASYNCcnt path), guarded --------------
#if __has_builtin(__builtin_amdgcn_global_load_async_to_lds_b128)
#define ASYNC_LDS 1
#endif

#ifdef ASYNC_LDS
// builtin expects pointers to 4 x i32 vectors: global (AS1) src, LDS (AS3) dst
typedef int vec4i __attribute__((vector_size(16)));
typedef __attribute__((address_space(1))) vec4i gas_v4i;
typedef __attribute__((address_space(3))) vec4i las_v4i;
#endif

// copy 16 bytes global -> LDS
__device__ __forceinline__ void copy16_g2l(const u16* g, u16* l) {
#ifdef ASYNC_LDS
  // generic LDS address carries the LDS byte offset in its low 32 bits
  __builtin_amdgcn_global_load_async_to_lds_b128(
      (gas_v4i*)(unsigned long long)g,
      (las_v4i*)(unsigned int)(unsigned long long)l,
      0, 0);
#else
  *reinterpret_cast<int4*>(l) = *reinterpret_cast<const int4*>(g);
#endif
}

#ifdef ASYNC_LDS
#if __has_builtin(__builtin_amdgcn_s_wait_asynccnt)
#define GEMM_WAIT(n) __builtin_amdgcn_s_wait_asynccnt(n)
#else
#define GEMM_WAIT(n) asm volatile("s_wait_asynccnt " #n ::: "memory")
#endif
#else
#define GEMM_WAIT(n) ((void)0)
#endif

// ---------------------------------------------------------------------------
__global__ void cvt_f32_bf16(const float* __restrict__ s, u16* __restrict__ d, int n) {
  for (int i = blockIdx.x * blockDim.x + threadIdx.x; i < n;
       i += gridDim.x * blockDim.x)
    d[i] = f2bf(s[i]);
}

// C[M,N] = A[M,K] * B[N,K]^T  (bf16 in, fp32 out)
// Block tile 128x128, 8 waves, wave tile 32x64 (8 WMMAs / K-step of 32).
// A/B panels (128x32 each, 8KB) double-buffered in LDS, filled by async
// global->LDS b128 copies (4 per thread per panel pair).
__global__ void __launch_bounds__(256) gemm_bf16_nt(
    const u16* __restrict__ A, const u16* __restrict__ B, float* __restrict__ C,
    int M, int N, int Kd) {
  __shared__ __align__(16) u16 sA[2][128 * 32];
  __shared__ __align__(16) u16 sB[2][128 * 32];
  const int tid  = threadIdx.x;
  const int lane = tid & 31;
  const int wid  = tid >> 5;
  const int m0 = blockIdx.y * 128;
  const int n0 = blockIdx.x * 128;
  const int wm = (wid >> 1) * 32;      // 0,32,64,96
  const int wn = (wid & 1) * 64;       // 0,64
  v8f acc0[4] = {}, acc1[4] = {};

  // stage one 128x32 A panel + 128x32 B panel into sA/sB[buf]
  auto stage = [&](int buf, int k0) {
#pragma unroll
    for (int c = tid; c < 512; c += 256) {       // 512 16B chunks per panel
      const int row = c >> 2, col = (c & 3) << 3;
      copy16_g2l(A + (size_t)(m0 + row) * Kd + k0 + col, &sA[buf][row * 32 + col]);
      copy16_g2l(B + (size_t)(n0 + row) * Kd + k0 + col, &sB[buf][row * 32 + col]);
    }
  };

  stage(0, 0);
  const int nk = Kd >> 5;
  for (int ks = 0; ks < nk; ++ks) {
    const int buf = ks & 1;
    if (ks + 1 < nk) {
      stage(buf ^ 1, (ks + 1) << 5);   // prefetch next panels (4 async ops)
      GEMM_WAIT(4);                    // current panels complete, next in flight
    } else {
      GEMM_WAIT(0);
    }
    __syncthreads();
    const u16* pa = sA[buf];
    const u16* pb = sB[buf];
    const v16bf a0 = load_frag_a(pa + (wm + 0) * 32, 32, lane);
    const v16bf a1 = load_frag_a(pa + (wm + 16) * 32, 32, lane);
#pragma unroll
    for (int j = 0; j < 4; ++j) {
      const v16bf b = load_frag_b(pb + (wn + j * 16) * 32, 32, lane);
      acc0[j] = wmma_bf16(a0, b, acc0[j]);
      acc1[j] = wmma_bf16(a1, b, acc1[j]);
    }
    __syncthreads();                   // protect buf from next stage's writes
  }
  // C/D layout: vgpr r, lanes 0-15 -> M=r, lanes 16-31 -> M=r+8; N = lane&15
  const int ncol = lane & 15, hi2 = lane >> 4;
#pragma unroll
  for (int j = 0; j < 4; ++j)
#pragma unroll
    for (int r = 0; r < 8; ++r) {
      const int mA = m0 + wm + hi2 * 8 + r;
      C[(size_t)mA * N + n0 + wn + j * 16 + ncol]        = acc0[j][r];
      C[(size_t)(mA + 16) * N + n0 + wn + j * 16 + ncol] = acc1[j][r];
    }
}

// RoPE on Q,K + repack to [H,T,64] bf16; V packed TRANSPOSED [Hkv,64,T] so
// PV B-fragments are key-contiguous b128 loads.
__global__ void rope_pack_kernel(const float* __restrict__ Qf,
                                 const float* __restrict__ Kf,
                                 const float* __restrict__ Vf,
                                 u16* __restrict__ Qb, u16* __restrict__ Kb,
                                 u16* __restrict__ Vtb) {
  const int t = blockIdx.x;
  const float ft = (float)t;
  for (int c = threadIdx.x; c < DM + 2 * KVD; c += blockDim.x) {
    if (c < DM) {
      const int h = c >> 6, d = c & 63, i = d & 31;
      const float ang = ft * __expf(-0.2878231366f * (float)i); // 10000^(-i/32)
      float sn, cs; __sincosf(ang, &sn, &cs);
      const float v  = Qf[t * DM + c];
      const float vp = Qf[t * DM + (h << 6) + ((d < 32) ? d + 32 : d - 32)];
      const float rot = (d < 32) ? -vp : vp;
      Qb[((size_t)h * T_SEQ + t) * HD + d] = f2bf(v * cs + rot * sn);
    } else if (c < DM + KVD) {
      const int cc = c - DM;
      const int h = cc >> 6, d = cc & 63, i = d & 31;
      const float ang = ft * __expf(-0.2878231366f * (float)i);
      float sn, cs; __sincosf(ang, &sn, &cs);
      const float v  = Kf[t * KVD + cc];
      const float vp = Kf[t * KVD + (h << 6) + ((d < 32) ? d + 32 : d - 32)];
      const float rot = (d < 32) ? -vp : vp;
      Kb[((size_t)h * T_SEQ + t) * HD + d] = f2bf(v * cs + rot * sn);
    } else {
      const int cc = c - DM - KVD;
      const int h = cc >> 6, d = cc & 63;
      Vtb[((size_t)h * HD + d) * T_SEQ + t] = f2bf(Vf[t * KVD + cc]);
    }
  }
}

// Flash-style sliding-window attention. One wave per (16-query tile, head).
// Per 32-key step: QK^T (4 WMMAs), online softmax (shfl_xor row reductions),
// P staged through LDS into A-fragment layout, PV (4 WMMAs).
__global__ void __launch_bounds__(32) gqa_attn_kernel(
    const u16* __restrict__ Q, const u16* __restrict__ K,
    const u16* __restrict__ Vt, u16* __restrict__ O) {
  __shared__ __align__(16) u16 ldsP[16 * 32];
  const int lane = threadIdx.x;
  const int h  = blockIdx.y;
  const int hk = h >> 2;                 // GQA group of 4
  const int q0 = blockIdx.x * 16;
  const u16* Qh = Q + ((size_t)h * T_SEQ + q0) * HD;
  const v16bf qa0 = load_frag_a(Qh, HD, lane);       // K-dim 0..31
  const v16bf qa1 = load_frag_a(Qh + 32, HD, lane);  // K-dim 32..63
  v8f o0 = {}, o1 = {}, o2 = {}, o3 = {};
  float rm[8], rl[8];
#pragma unroll
  for (int r = 0; r < 8; ++r) { rm[r] = -1e30f; rl[r] = 0.f; }
  const int ncol = lane & 15, hi = lane >> 4;
  int jbeg = q0 - HALF_WIN; if (jbeg < 0) jbeg = 0; jbeg &= ~31;
  int jend = q0 + 16 + HALF_WIN; if (jend > T_SEQ) jend = T_SEQ;
  jend = (jend + 31) & ~31;
  const float scale = 0.125f;            // 1/sqrt(64)
  for (int j0 = jbeg; j0 < jend; j0 += 32) {
    const u16* Kh = K + ((size_t)hk * T_SEQ + j0) * HD;
    v8f z = {};
    v8f s0 = wmma_bf16(qa1, load_frag_b(Kh + 32, HD, lane),
             wmma_bf16(qa0, load_frag_b(Kh, HD, lane), z));
    v8f s1 = wmma_bf16(qa1, load_frag_b(Kh + 16 * HD + 32, HD, lane),
             wmma_bf16(qa0, load_frag_b(Kh + 16 * HD, HD, lane), z));
#pragma unroll
    for (int r = 0; r < 8; ++r) {
      const int m  = q0 + hi * 8 + r;    // global query row for this (r, half)
      const int jA = j0 + ncol, jB = jA + 16;
      const bool aA = (jA >= m - HALF_WIN) && (jA < m + HALF_WIN);
      const bool aB = (jB >= m - HALF_WIN) && (jB < m + HALF_WIN);
      const float sA = aA ? s0[r] * scale : -1e30f;
      const float sB = aB ? s1[r] * scale : -1e30f;
      float t = fmaxf(sA, sB);
#pragma unroll
      for (int off = 8; off; off >>= 1) t = fmaxf(t, __shfl_xor(t, off, 16));
      const float mn = fmaxf(rm[r], t);
      const float c  = __expf(rm[r] - mn);
      const float pA = aA ? __expf(sA - mn) : 0.f;
      const float pB = aB ? __expf(sB - mn) : 0.f;
      float ps = pA + pB;
#pragma unroll
      for (int off = 8; off; off >>= 1) ps += __shfl_xor(ps, off, 16);
      rl[r] = rl[r] * c + ps;
      rm[r] = mn;
      o0[r] *= c; o1[r] *= c; o2[r] *= c; o3[r] *= c;
      ldsP[(hi * 8 + r) * 32 + ncol]      = f2bf(pA);
      ldsP[(hi * 8 + r) * 32 + 16 + ncol] = f2bf(pB);
    }
    __syncthreads();                     // order LDS stores before frag reads
    const v16bf pf = load_frag_a(ldsP, 32, lane);
    const u16* Vh = Vt + (size_t)hk * HD * T_SEQ + j0;  // [Hkv,64,T]
    o0 = wmma_bf16(pf, load_frag_b(Vh +  0 * T_SEQ, T_SEQ, lane), o0);
    o1 = wmma_bf16(pf, load_frag_b(Vh + 16 * T_SEQ, T_SEQ, lane), o1);
    o2 = wmma_bf16(pf, load_frag_b(Vh + 32 * T_SEQ, T_SEQ, lane), o2);
    o3 = wmma_bf16(pf, load_frag_b(Vh + 48 * T_SEQ, T_SEQ, lane), o3);
    __syncthreads();
  }
#pragma unroll
  for (int r = 0; r < 8; ++r) {
    const int t = q0 + hi * 8 + r;
    const float inv = 1.f / rl[r];
    u16* dst = O + (size_t)t * DM + h * HD + ncol;
    dst[0]  = f2bf(o0[r] * inv);
    dst[16] = f2bf(o1[r] * inv);
    dst[32] = f2bf(o2[r] * inv);
    dst[48] = f2bf(o3[r] * inv);
  }
}

// ---------------------------------------------------------------------------
extern "C" void kernel_launch(void* const* d_in, const int* in_sizes, int n_in,
                              void* d_out, int out_size, void* d_ws, size_t ws_size,
                              hipStream_t stream) {
  (void)in_sizes; (void)n_in; (void)out_size; (void)ws_size;
  const float* hs = (const float*)d_in[0];
  const float* Wq = (const float*)d_in[1];
  const float* Wk = (const float*)d_in[2];
  const float* Wv = (const float*)d_in[3];
  const float* Wo = (const float*)d_in[4];
  float* out = (float*)d_out;

  char* w = (char*)d_ws;
  u16* hs_b = (u16*)w;  w += (size_t)T_SEQ * DM * 2;     // 8 MiB
  u16* wq_b = (u16*)w;  w += (size_t)DM * DM * 2;        // 8 MiB
  u16* wk_b = (u16*)w;  w += (size_t)KVD * DM * 2;       // 2 MiB
  u16* wv_b = (u16*)w;  w += (size_t)KVD * DM * 2;       // 2 MiB
  u16* wo_b = (u16*)w;  w += (size_t)DM * DM * 2;        // 8 MiB
  float* qf = (float*)w; w += (size_t)T_SEQ * DM * 4;    // 16 MiB
  float* kf = (float*)w; w += (size_t)T_SEQ * KVD * 4;   // 4 MiB
  float* vf = (float*)w; w += (size_t)T_SEQ * KVD * 4;   // 4 MiB
  u16* qb  = (u16*)w;  w += (size_t)NH  * T_SEQ * HD * 2;  // 8 MiB  [H,T,64]
  u16* kb  = (u16*)w;  w += (size_t)NKV * T_SEQ * HD * 2;  // 2 MiB  [Hkv,T,64]
  u16* vtb = (u16*)w;  w += (size_t)NKV * HD * T_SEQ * 2;  // 2 MiB  [Hkv,64,T]
  u16* ab  = (u16*)w;  w += (size_t)T_SEQ * DM * 2;        // 8 MiB  attn out

  // 1) fp32 -> bf16 operand conversion
  cvt_f32_bf16<<<2048, 256, 0, stream>>>(hs, hs_b, T_SEQ * DM);
  cvt_f32_bf16<<<2048, 256, 0, stream>>>(Wq, wq_b, DM * DM);
  cvt_f32_bf16<<<1024, 256, 0, stream>>>(Wk, wk_b, KVD * DM);
  cvt_f32_bf16<<<1024, 256, 0, stream>>>(Wv, wv_b, KVD * DM);
  cvt_f32_bf16<<<2048, 256, 0, stream>>>(Wo, wo_b, DM * DM);

  // 2) projections: y = x @ W^T
  gemm_bf16_nt<<<dim3(DM / 128, T_SEQ / 128), 256, 0, stream>>>(hs_b, wq_b, qf, T_SEQ, DM, DM);
  gemm_bf16_nt<<<dim3(KVD / 128, T_SEQ / 128), 256, 0, stream>>>(hs_b, wk_b, kf, T_SEQ, KVD, DM);
  gemm_bf16_nt<<<dim3(KVD / 128, T_SEQ / 128), 256, 0, stream>>>(hs_b, wv_b, vf, T_SEQ, KVD, DM);

  // 3) RoPE + repack (V transposed)
  rope_pack_kernel<<<T_SEQ, 256, 0, stream>>>(qf, kf, vf, qb, kb, vtb);

  // 4) sliding-window flash attention
  gqa_attn_kernel<<<dim3(T_SEQ / 16, NH), 32, 0, stream>>>(qb, kb, vtb, ab);

  // 5) output projection -> fp32 d_out
  gemm_bf16_nt<<<dim3(DM / 128, T_SEQ / 128), 256, 0, stream>>>(ab, wo_b, out, T_SEQ, DM, DM);
}
